// DU_GSB_LSTM_68264210202916
// MI455X (gfx1250) — matrix operand order repaired and checked by reading, not compiled, not run on hardware
//
#include <hip/hip_runtime.h>
#include <hip/hip_bf16.h>
#include <math.h>

#define NDIM 1024   // Ising dimension (rb*N)
#define BATCH 1024
#define NHT 512     // 2*Nr
#define TSTEPS 32

typedef __attribute__((ext_vector_type(16))) __bf16 v16bf;
typedef __attribute__((ext_vector_type(8)))  float  v8f;
typedef __attribute__((ext_vector_type(4)))  float  f4;
typedef __attribute__((ext_vector_type(4)))  int    i4;

__device__ __forceinline__ float sigf(float z){ return 1.0f/(1.0f+__expf(-z)); }
__device__ __forceinline__ float phi_s(float x){
  float a=x+1.0f, b=x-1.0f;
  return a*sigf(a) - b*sigf(b) - 1.0f;
}
__device__ __forceinline__ float psi_s(float x){ return sigf(100.0f*(fabsf(x)-1.01f)); }
__device__ __forceinline__ float rng01(unsigned a, unsigned b){
  unsigned h = a*0x9E3779B9u ^ b*0x85EBCA6Bu;
  h ^= h>>16; h *= 0x7feb352du; h ^= h>>15; h *= 0x846ca68bu; h ^= h>>16;
  return (float)h * 2.3283064365386963e-10f;
}

// ---------------- setup kernels ----------------

__global__ void k_scal_init(float* scal){
  scal[0] = INFINITY;  // gbest_e
  scal[1] = 0.0f;      // ||J||_F^2 accumulator
  scal[2] = 0.0f;      // c0_coef
  scal[3]=scal[4]=scal[5]=0.0f;   // fmean
  scal[6]=scal[7]=scal[8]=1.0f;   // fstd
}

__global__ void k_build_ht(const float* __restrict__ Hr, const float* __restrict__ Hi,
                           const float* __restrict__ yr, const float* __restrict__ yi,
                           float* __restrict__ Ht, float* __restrict__ yt){
  int j = blockIdx.x*16 + threadIdx.x;
  int i = blockIdx.y*16 + threadIdx.y;
  float v;
  if (i < 256) v = (j < 256) ? Hr[i*256+j] : -Hi[i*256+(j-256)];
  else { int i2=i-256; v = (j < 256) ? Hi[i2*256+j] : Hr[i2*256+(j-256)]; }
  Ht[i*NHT+j] = v;
  if (j == 0) yt[i] = (i < 256) ? yr[i] : yi[i-256];
}

// AUG = [Ht*Ht^T + lam*I | I]   (512 x 1024)
__global__ void k_gram(const float* __restrict__ Ht, float* __restrict__ AUG,
                       const float* __restrict__ lam){
  int j = blockIdx.x*16 + threadIdx.x;
  int i = blockIdx.y*16 + threadIdx.y;
  float acc = 0.f;
  for (int k = 0; k < NHT; ++k) acc += Ht[i*NHT+k]*Ht[j*NHT+k];
  if (i == j) acc += lam[0];
  AUG[i*1024 + j] = acc;
  AUG[i*1024 + 512 + j] = (i == j) ? 1.0f : 0.0f;
}

// Gauss-Jordan inverse of the SPD 512x512 block (single workgroup, 1024 threads)
__global__ void __launch_bounds__(1024) k_ginv(float* __restrict__ AUG){
  __shared__ float fac[NHT];
  __shared__ float pivinv;
  int tid = threadIdx.x;
  for (int k = 0; k < NHT; ++k){
    if (tid == 0) pivinv = 1.0f / AUG[k*1024 + k];
    __syncthreads();
    AUG[k*1024 + tid] *= pivinv;        // 1024 cols, one per thread
    __syncthreads();
    if (tid < NHT) fac[tid] = AUG[tid*1024 + k];
    __syncthreads();
    for (int idx = tid; idx < NHT*1024; idx += 1024){
      int r = idx >> 10, j = idx & 1023;
      if (r != k) AUG[idx] -= fac[r] * AUG[k*1024 + j];
    }
    __syncthreads();
  }
}

// UHt = (A^-1/lam) @ Ht
__global__ void k_uht(const float* __restrict__ AUG, const float* __restrict__ Ht,
                      float* __restrict__ UHt, const float* __restrict__ lam){
  int j = blockIdx.x*16 + threadIdx.x;
  int i = blockIdx.y*16 + threadIdx.y;
  float acc = 0.f;
  for (int k = 0; k < NHT; ++k) acc += AUG[i*1024 + 512 + k]*Ht[k*NHT + j];
  UHt[i*NHT + j] = acc / lam[0];
}

// G = Ht^T @ UHt
__global__ void k_g(const float* __restrict__ Ht, const float* __restrict__ UHt,
                    float* __restrict__ G){
  int c2 = blockIdx.x*16 + threadIdx.x;
  int c1 = blockIdx.y*16 + threadIdx.y;
  float acc = 0.f;
  for (int i2 = 0; i2 < NHT; ++i2) acc += Ht[i2*NHT + c1]*UHt[i2*NHT + c2];
  G[c1*NHT + c2] = acc;
}

// J[m1][m2] = (m1==m2)?0 : -(2/qv)*s(r1)*s(r2)*G[c1][c2]  (bf16) + Frobenius accum
__global__ void __launch_bounds__(256) k_jpack(const float* __restrict__ G,
                                               __bf16* __restrict__ Jb,
                                               float* __restrict__ scal){
  __shared__ float red[256];
  int m1 = blockIdx.x, tid = threadIdx.x;
  int r1 = m1 >> 9, c1 = m1 & 511;
  float s1v = (r1 == 0) ? 2.0f : 1.0f;
  const float qv = 10.0f;  // 2*(16-1)/3
  float ss = 0.f;
  for (int m2 = tid; m2 < NDIM; m2 += 256){
    int r2 = m2 >> 9, c2 = m2 & 511;
    float s2v = (r2 == 0) ? 2.0f : 1.0f;
    float jv = (m1 == m2) ? 0.0f : -(2.0f/qv)*s1v*s2v*G[c1*NHT + c2];
    Jb[(size_t)m1*NDIM + m2] = (__bf16)jv;
    ss += jv*jv;
  }
  red[tid] = ss; __syncthreads();
  for (int s = 128; s > 0; s >>= 1){ if (tid < s) red[tid] += red[tid+s]; __syncthreads(); }
  if (tid == 0) atomicAdd(&scal[1], red[0]);
}

__global__ void k_scal_final(float* scal){
  scal[2] = 0.5f*sqrtf(1023.0f)/(sqrtf(scal[1]) + 1e-12f);  // c0_coef
}

// h vector (single block of 512)
__global__ void __launch_bounds__(512) k_hvec(const float* __restrict__ Ht,
      const float* __restrict__ AUG, const float* __restrict__ yt,
      const float* __restrict__ lam, float* __restrict__ Htones,
      float* __restrict__ zv, float* __restrict__ Uz, float* __restrict__ Hv,
      float* __restrict__ hvec){
  int tid = threadIdx.x;
  float a = 0.f;
  for (int j = 0; j < NHT; ++j) a += Ht[tid*NHT + j];
  Htones[tid] = a;
  const float sqrtM = 4.0f, ssum = 3.0f, qv = 10.0f;
  zv[tid] = (yt[tid] + (sqrtM - 1.0f - ssum)*a) / sqrtf(qv);
  __threadfence_block(); __syncthreads();
  float u = 0.f;
  for (int k = 0; k < NHT; ++k) u += AUG[tid*1024 + 512 + k]*zv[k];
  Uz[tid] = u / lam[0];
  __threadfence_block(); __syncthreads();
  float hc = 0.f;
  for (int i2 = 0; i2 < NHT; ++i2) hc += Ht[i2*NHT + tid]*Uz[i2];
  Hv[tid] = hc;
  __threadfence_block(); __syncthreads();
  for (int m = tid; m < NDIM; m += NHT){
    int r = m >> 9, c = m & 511;
    hvec[m] = 2.0f*((r == 0) ? 2.0f : 1.0f)*Hv[c];
  }
}

__global__ void k_init_state(const float* __restrict__ x0, const float* __restrict__ y0,
                             float* __restrict__ XT, float* __restrict__ YT,
                             float* __restrict__ VM){
  size_t idx = (size_t)blockIdx.x*256 + threadIdx.x;
  int b = (int)(idx >> 10), n = (int)(idx & 1023);
  XT[idx] = 0.02f*(x0[(size_t)n*BATCH + b] - 0.5f);
  YT[idx] = 0.02f*(y0[(size_t)n*BATCH + b] - 0.5f);
  VM[idx] = 0.0f;
}

__global__ void k_init_lstm(const float* __restrict__ h0, const float* __restrict__ c0,
                            float* __restrict__ hid, float* __restrict__ cel){
  int b = blockIdx.x, t = threadIdx.x;
  hid[(size_t)b*64 + t] = h0[t];
  cel[(size_t)b*64 + t] = c0[t];
}

// ---------------- per-step kernels ----------------

// S = phi(x) (zero->1) and X, both to bf16 panel [S ; X] of shape [2048][1024]
__global__ void k_pack(const float* __restrict__ XT, __bf16* __restrict__ XS){
  size_t idx = (size_t)blockIdx.x*256 + threadIdx.x;
  float x = XT[idx];
  float s = phi_s(x);
  if (s == 0.0f) s = 1.0f;
  XS[idx] = (__bf16)s;
  XS[(size_t)NDIM*BATCH + idx] = (__bf16)x;
}

// Out[bcol][m] = sum_k XS[bcol][k] * J[m][k]   (M=1024, N=2048, K=1024), bf16 WMMA.
// Software-pipelined: loads for step k+1 are issued before the WMMAs of step k
// (ping-pong register fragments, fully unrolled so indices are compile-time).
__global__ void __launch_bounds__(256) k_gemm(const __bf16* __restrict__ Jb,
                                              const __bf16* __restrict__ XS,
                                              float* __restrict__ Out){
  const int lane = threadIdx.x & 31;
  const int wave = threadIdx.x >> 5;
  const int task = blockIdx.x*8 + wave;      // 2048 tasks
  const int mtile = task & 63;               // 64 m-tiles
  const int bsup  = task >> 6;               // 32 supertiles of 64 batch cols
  const int hi    = lane >> 4;
  const int l15   = lane & 15;
  const int arow  = mtile*16 + l15;          // A: 16x32, lane holds one J row
  const int akb   = hi ? 16 : 0;             // A lanes>=16: K = 16..31
  const int bkb   = hi ? 16 : 0;             // B lanes>=16: K = 16..31

  union U32 { i4 q[2]; v16bf v; };
  v8f zero = {0.f,0.f,0.f,0.f,0.f,0.f,0.f,0.f};
  v8f acc[4];
#pragma unroll
  for (int t = 0; t < 4; ++t) acc[t] = zero;

  const __bf16* aBase = Jb + (size_t)arow*NDIM + akb;
  const __bf16* bBase[4];
#pragma unroll
  for (int t = 0; t < 4; ++t)
    bBase[t] = XS + (size_t)(bsup*64 + t*16 + l15)*NDIM + bkb;

  U32 A[2]; U32 B[2][4];

  // prologue: fragments for kt = 0
  A[0].q[0] = *(const i4*)(aBase);
  A[0].q[1] = *(const i4*)(aBase + 8);
#pragma unroll
  for (int t = 0; t < 4; ++t){
    B[0][t].q[0] = *(const i4*)(bBase[t]);
    B[0][t].q[1] = *(const i4*)(bBase[t] + 8);
  }

#pragma unroll
  for (int kt = 0; kt < 31; ++kt){
    const int cur = kt & 1, nxt = cur ^ 1;
    // issue next iteration's loads first (overlap with WMMAs below)
    const __bf16* ap = aBase + (kt+1)*32;
    A[nxt].q[0] = *(const i4*)(ap);
    A[nxt].q[1] = *(const i4*)(ap + 8);
#pragma unroll
    for (int t = 0; t < 4; ++t){
      const __bf16* bp = bBase[t] + (kt+1)*32;
      B[nxt][t].q[0] = *(const i4*)(bp);
      B[nxt][t].q[1] = *(const i4*)(bp + 8);
    }
    if ((kt & 7) == 0 && kt + 8 < 32)
      __builtin_prefetch(aBase + (kt+8)*32, 0, 1);   // global_prefetch on hot J stream
#pragma unroll
    for (int t = 0; t < 4; ++t)
      acc[t] = __builtin_amdgcn_wmma_f32_16x16x32_bf16(
          false, A[cur].v, false, B[cur][t].v, (short)0, acc[t], false, false);
  }
  // epilogue: kt = 31 uses buffer 1
#pragma unroll
  for (int t = 0; t < 4; ++t)
    acc[t] = __builtin_amdgcn_wmma_f32_16x16x32_bf16(
        false, A[1].v, false, B[1][t].v, (short)0, acc[t], false, false);

  const int mrow = mtile*16 + (hi ? 8 : 0);  // D: lane holds 8 consecutive rows
#pragma unroll
  for (int t = 0; t < 4; ++t){
    const int bcol = bsup*64 + t*16 + l15;
    union { v8f v; f4 f[2]; } R; R.v = acc[t];
    float* op = Out + (size_t)bcol*NDIM + mrow;
    *(f4*)(op)     = R.f[0];
    *(f4*)(op + 4) = R.f[1];
  }
}

__global__ void __launch_bounds__(256) k_energy(const float* __restrict__ XT,
      const float* __restrict__ AsT, const float* __restrict__ hv,
      float* __restrict__ e, float* __restrict__ sx, float* __restrict__ sxx){
  __shared__ f4 red[256];
  int b = blockIdx.x, tid = threadIdx.x;
  float a1=0,a2=0,a3=0,a4=0;
  for (int n = tid; n < NDIM; n += 256){
    float x = XT[(size_t)b*NDIM + n];
    float s = phi_s(x); if (s == 0.f) s = 1.f;
    float As = AsT[(size_t)b*NDIM + n];
    a1 += s*As; a2 += hv[n]*s; a3 += x; a4 += x*x;
  }
  f4 r = {a1,a2,a3,a4};
  red[tid] = r; __syncthreads();
  for (int s = 128; s > 0; s >>= 1){
    if (tid < s){ f4 o = red[tid+s]; red[tid] += o; }
    __syncthreads();
  }
  if (tid == 0){ f4 t = red[0]; e[b] = -0.5f*t.x - t.y; sx[b] = t.z; sxx[b] = t.w; }
}

__global__ void __launch_bounds__(1024) k_argmin(const float* __restrict__ e,
      const float* __restrict__ XT, float* __restrict__ gbest, float* __restrict__ scal){
  __shared__ float v[1024]; __shared__ int ix[1024];
  __shared__ int bestIdx; __shared__ int doUpd;
  int tid = threadIdx.x;
  v[tid] = e[tid]; ix[tid] = tid;
  __syncthreads();
  for (int s = 512; s > 0; s >>= 1){
    if (tid < s){
      float vo = v[tid+s]; int io = ix[tid+s];
      if (vo < v[tid] || (vo == v[tid] && io < ix[tid])){ v[tid] = vo; ix[tid] = io; }
    }
    __syncthreads();
  }
  if (tid == 0){
    if (v[0] < scal[0]){ scal[0] = v[0]; bestIdx = ix[0]; doUpd = 1; } else doUpd = 0;
  }
  __syncthreads();
  if (doUpd) gbest[tid] = XT[(size_t)bestIdx*NDIM + tid];
}

__global__ void __launch_bounds__(256) k_feats(const float* __restrict__ XT,
      const float* __restrict__ YT, const float* __restrict__ gbest,
      const float* __restrict__ e, const float* __restrict__ sx, const float* __restrict__ sxx,
      float* __restrict__ feats, float* __restrict__ dotv, float* __restrict__ yn2){
  __shared__ f4 red[256];
  int b = blockIdx.x, tid = threadIdx.x;
  float g2=0, dt=0, y2=0;
  for (int n = tid; n < NDIM; n += 256){
    float x  = XT[(size_t)b*NDIM + n];
    float yv = YT[(size_t)b*NDIM + n];
    float d  = gbest[n] - x;
    g2 += d*d; dt += d*yv; y2 += yv*yv;
  }
  f4 r = {g2,dt,y2,0.f};
  red[tid] = r; __syncthreads();
  for (int s = 128; s > 0; s >>= 1){
    if (tid < s){ f4 o = red[tid+s]; red[tid] += o; }
    __syncthreads();
  }
  if (tid == 0){
    f4 t = red[0];
    feats[b*3+0] = e[b];
    feats[b*3+1] = sqrtf(t.x);
    float mean = sx[b]/(float)NDIM;
    float var  = (sxx[b] - sx[b]*mean)/(float)(NDIM-1);
    feats[b*3+2] = sqrtf(fmaxf(var, 0.f));
    dotv[b] = t.y; yn2[b] = t.z;
  }
}

__global__ void __launch_bounds__(256) k_fstats(const float* __restrict__ feats,
                                                float* __restrict__ scal){
  __shared__ float s1[3][256]; __shared__ float s2[3][256];
  int tid = threadIdx.x;
  float m[3]={0,0,0}, q[3]={0,0,0};
  for (int b = tid; b < BATCH; b += 256)
    for (int f = 0; f < 3; ++f){ float v = feats[b*3+f]; m[f] += v; q[f] += v*v; }
  for (int f = 0; f < 3; ++f){ s1[f][tid] = m[f]; s2[f][tid] = q[f]; }
  __syncthreads();
  for (int s = 128; s > 0; s >>= 1){
    if (tid < s) for (int f = 0; f < 3; ++f){ s1[f][tid] += s1[f][tid+s]; s2[f][tid] += s2[f][tid+s]; }
    __syncthreads();
  }
  if (tid < 3){
    float mean = s1[tid][0]/(float)BATCH;
    float var  = (s2[tid][0] - s1[tid][0]*mean)/(float)(BATCH-1);
    scal[3+tid] = mean;
    scal[6+tid] = fmaxf(sqrtf(fmaxf(var, 0.f)), 1e-6f);
  }
}

__global__ void __launch_bounds__(256) k_lstm(const float* __restrict__ feats,
      const float* __restrict__ scal,
      const float* __restrict__ W_ih, const float* __restrict__ b_ih,
      const float* __restrict__ W_hh, const float* __restrict__ b_hh,
      const float* __restrict__ W1,  const float* __restrict__ b1,
      const float* __restrict__ W2,  const float* __restrict__ b2,
      float* __restrict__ hid, float* __restrict__ cel,
      float* __restrict__ aarr, float* __restrict__ barr){
  __shared__ float sh[64]; __shared__ float sg[256]; __shared__ float sr[64]; __shared__ float sab[2];
  int b = blockIdx.x, tid = threadIdx.x;
  float nf[3];
  for (int f = 0; f < 3; ++f) nf[f] = (feats[b*3+f] - scal[3+f]) / scal[6+f];
  if (tid < 64) sh[tid] = hid[(size_t)b*64 + tid];
  __syncthreads();
  float g = b_ih[tid] + b_hh[tid];
  for (int j = 0; j < 3;  ++j) g += nf[j]*W_ih[tid*3 + j];
  for (int j = 0; j < 64; ++j) g += sh[j]*W_hh[tid*64 + j];
  sg[tid] = g;
  __syncthreads();
  if (tid < 64){
    float gi = sigf(sg[tid]), gf = sigf(sg[64+tid]);
    float gg = tanhf(sg[128+tid]), go = sigf(sg[192+tid]);
    float cn = gf*cel[(size_t)b*64 + tid] + gi*gg;
    float hn = go*tanhf(cn);
    cel[(size_t)b*64 + tid] = cn; hid[(size_t)b*64 + tid] = hn;
    sh[tid] = hn;
  }
  __syncthreads();
  if (tid < 64){
    float r = b1[tid];
    for (int j = 0; j < 64; ++j) r += sh[j]*W1[tid*64 + j];
    sr[tid] = fmaxf(r, 0.f);
  }
  __syncthreads();
  if (tid < 2){
    float z = b2[tid];
    for (int j = 0; j < 64; ++j) z += sr[j]*W2[tid*64 + j];
    float sp = (z > 20.f) ? z : log1pf(__expf(z));
    sab[tid] = sp - 1.0f;
  }
  __syncthreads();
  if (tid == 0){ aarr[b] = sab[0]; barr[b] = sab[1]; }
}

__global__ void __launch_bounds__(256) k_update(float* __restrict__ XT, float* __restrict__ YT,
      float* __restrict__ VM, const float* __restrict__ AxT, const float* __restrict__ hv,
      const float* __restrict__ gbest, const float* __restrict__ feats,
      const float* __restrict__ dotv, const float* __restrict__ yn2,
      const float* __restrict__ aarr, const float* __restrict__ barr,
      const float* __restrict__ scal, const float* __restrict__ delta,
      const float* __restrict__ eta, int i, float p, float stepf){
  size_t idx = (size_t)blockIdx.x*256 + threadIdx.x;
  int b = (int)(idx >> 10), n = (int)(idx & 1023);
  float x = XT[idx], yv = YT[idx], vm = VM[idx];
  float gnorm = feats[b*3+1];
  float f = (1.0f - stepf)/(1.0f + gnorm);
  const float eps = 1e-8f;
  float cosv = (f*dotv[b]) / (fmaxf(f*gnorm, eps)*fmaxf(sqrtf(yn2[b]), eps));
  float mask = tanhf(1000.0f*(cosv + barr[b]));
  float ist  = mask * f * (gbest[n] - x);
  float alpha = 0.9f - 0.4f*stepf;
  float vmn = alpha*vm + (1.0f - alpha)*ist;
  float di = delta[i], ei = eta[i], c0c = scal[2];
  float yvn = yv + (-(1.0f - p)*x + ei*c0c*(AxT[idx] + hv[n]) + vmn)*di;
  float xn = x + di*yvn;
  float r = rng01((unsigned)(i*1315423911u + 2654435761u), (unsigned)idx);
  if (r < p) xn -= aarr[b];
  xn = phi_s(xn);
  yvn *= (1.0f - psi_s(xn));
  XT[idx] = xn; YT[idx] = yvn; VM[idx] = vmn;
}

__global__ void k_copy(const float* __restrict__ XT, float* __restrict__ out){
  size_t idx = (size_t)blockIdx.x*256 + threadIdx.x;
  out[idx] = XT[idx];
}

// ---------------- host ----------------

extern "C" void kernel_launch(void* const* d_in, const int* in_sizes, int n_in,
                              void* d_out, int out_size, void* d_ws, size_t ws_size,
                              hipStream_t stream){
  const float* Hr   = (const float*)d_in[0];
  const float* Hi   = (const float*)d_in[1];
  const float* yr   = (const float*)d_in[2];
  const float* yi   = (const float*)d_in[3];
  const float* x0   = (const float*)d_in[4];
  const float* y0   = (const float*)d_in[5];
  const float* lam  = (const float*)d_in[6];
  const float* delta= (const float*)d_in[7];
  const float* eta  = (const float*)d_in[8];
  const float* W_ih = (const float*)d_in[9];
  const float* W_hh = (const float*)d_in[10];
  const float* b_ih = (const float*)d_in[11];
  const float* b_hh = (const float*)d_in[12];
  const float* W1   = (const float*)d_in[13];
  const float* b1   = (const float*)d_in[14];
  const float* W2   = (const float*)d_in[15];
  const float* b2   = (const float*)d_in[16];
  const float* h0   = (const float*)d_in[17];
  const float* c0   = (const float*)d_in[18];

  char* w = (char*)d_ws;
  auto bump = [&](size_t bytes)->char*{
    char* p = w; w += (bytes + 255) & ~(size_t)255; return p;
  };
  float*  XT    = (float*) bump((size_t)BATCH*NDIM*4);
  float*  YT    = (float*) bump((size_t)BATCH*NDIM*4);
  float*  VM    = (float*) bump((size_t)BATCH*NDIM*4);
  float*  OUT2  = (float*) bump((size_t)2*BATCH*NDIM*4);  // [AsT ; AxT]
  __bf16* XSbf  = (__bf16*)bump((size_t)2*BATCH*NDIM*2);  // [S ; X]
  __bf16* Jbf   = (__bf16*)bump((size_t)NDIM*NDIM*2);
  float*  Ht    = (float*) bump((size_t)NHT*NHT*4);
  float*  AUG   = (float*) bump((size_t)NHT*1024*4);
  float*  UHt   = (float*) bump((size_t)NHT*NHT*4);
  float*  G     = (float*) bump((size_t)NHT*NHT*4);
  float*  hvec  = (float*) bump(NDIM*4);
  float*  yt    = (float*) bump(NHT*4);
  float*  zv    = (float*) bump(NHT*4);
  float*  Htones= (float*) bump(NHT*4);
  float*  Uz    = (float*) bump(NHT*4);
  float*  Hv    = (float*) bump(NHT*4);
  float*  e     = (float*) bump(BATCH*4);
  float*  sx    = (float*) bump(BATCH*4);
  float*  sxx   = (float*) bump(BATCH*4);
  float*  feats = (float*) bump(BATCH*3*4);
  float*  dotv  = (float*) bump(BATCH*4);
  float*  yn2   = (float*) bump(BATCH*4);
  float*  aarr  = (float*) bump(BATCH*4);
  float*  barr  = (float*) bump(BATCH*4);
  float*  gbest = (float*) bump(NDIM*4);
  float*  scal  = (float*) bump(64*4);
  float*  hid   = (float*) bump((size_t)BATCH*64*4);
  float*  cel   = (float*) bump((size_t)BATCH*64*4);
  (void)ws_size; (void)in_sizes; (void)n_in; (void)out_size;

  const dim3 g2(32,32), b2d(16,16);
  const int EW = (BATCH*NDIM)/256;   // 4096 blocks for 1M-thread elementwise grids

  // ---- setup ----
  k_scal_init<<<1,1,0,stream>>>(scal);
  k_build_ht<<<g2,b2d,0,stream>>>(Hr,Hi,yr,yi,Ht,yt);
  k_gram<<<g2,b2d,0,stream>>>(Ht,AUG,lam);
  k_ginv<<<1,1024,0,stream>>>(AUG);
  k_uht<<<g2,b2d,0,stream>>>(AUG,Ht,UHt,lam);
  k_g<<<g2,b2d,0,stream>>>(Ht,UHt,G);
  k_jpack<<<NDIM,256,0,stream>>>(G,Jbf,scal);
  k_scal_final<<<1,1,0,stream>>>(scal);
  k_hvec<<<1,512,0,stream>>>(Ht,AUG,yt,lam,Htones,zv,Uz,Hv,hvec);
  k_init_state<<<EW,256,0,stream>>>(x0,y0,XT,YT,VM);
  k_init_lstm<<<BATCH,64,0,stream>>>(h0,c0,hid,cel);

  // ---- main loop ----
  const float* AsT = OUT2;
  const float* AxT = OUT2 + (size_t)BATCH*NDIM;
  for (int i = 0; i < TSTEPS; ++i){
    float stepf = (float)i / (float)TSTEPS;
    float p = 0.1f + 0.45f*(1.0f + cosf(3.14159265358979f*stepf));
    k_pack<<<EW,256,0,stream>>>(XT,XSbf);
    k_gemm<<<256,256,0,stream>>>(Jbf,XSbf,OUT2);
    k_energy<<<BATCH,256,0,stream>>>(XT,AsT,hvec,e,sx,sxx);
    k_argmin<<<1,1024,0,stream>>>(e,XT,gbest,scal);
    k_feats<<<BATCH,256,0,stream>>>(XT,YT,gbest,e,sx,sxx,feats,dotv,yn2);
    if (i == 0) k_fstats<<<1,256,0,stream>>>(feats,scal);
    k_lstm<<<BATCH,256,0,stream>>>(feats,scal,W_ih,b_ih,W_hh,b_hh,W1,b1,W2,b2,
                                   hid,cel,aarr,barr);
    k_update<<<EW,256,0,stream>>>(XT,YT,VM,AxT,hvec,gbest,feats,dotv,yn2,
                                  aarr,barr,scal,delta,eta,i,p,stepf);
  }
  k_copy<<<EW,256,0,stream>>>(XT,(float*)d_out);
}
